// ReverseTLSTMLayer_45226005627005
// MI455X (gfx1250) — matrix operand order repaired and checked by reading, not compiled, not run on hardware
//
#include <hip/hip_runtime.h>
#include <hip/hip_bf16.h>
#include <math.h>

// Problem dims (fixed by the reference)
#define T_LEN 512
#define BATCH 64
#define DIM   1024
#define HID   1024
#define NG    (4 * HID)   // 4096 gate columns

typedef __bf16 bf16_t;
typedef __attribute__((ext_vector_type(16))) __bf16 v16bf;
typedef __attribute__((ext_vector_type(8)))  __bf16 v8bf;
typedef __attribute__((ext_vector_type(8)))  float  v8f;

// ---------------------------------------------------------------------------
// Fragment loaders; C/D per CDNA5 ISA (CDNA5 ISA §7.12.2), A/B use this code's contiguous K order, not the ISA table order
// A (16-bit, 16x32 MxK): lane<16 -> row=lane, K {0..15};
//                        lane>=16 -> row=lane-16, K {16..31}
// B (16-bit, 32x16 KxN): lane -> col=lane&15, K contiguous {0..15} / {16..31}
// C/D (f32 16x16): VGPR r -> row r (lanes 0-15) or r+8 (lanes 16-31), col=lane&15
// ---------------------------------------------------------------------------

__device__ __forceinline__ v16bf load_a_bf16(const bf16_t* __restrict__ row, int lane) {
  const int kb = (lane < 16) ? 0 : 16;
  v8bf lo = *(const v8bf*)(row + kb);        // K kb..kb+7     (16B aligned)
  v8bf hi = *(const v8bf*)(row + kb + 8);    // K kb+8..kb+15
  v16bf r;
#pragma unroll
  for (int i = 0; i < 8; ++i) { r[i] = lo[i]; r[8 + i] = hi[i]; }
  return r;
}

// B-fragment: W is row-major [N, K]; lane's 16 K-values are contiguous in W's row.
__device__ __forceinline__ v16bf load_b_bf16(const bf16_t* __restrict__ W, int n0,
                                             size_t ldk, int k, int lane) {
  const int n  = n0 + (lane & 15);
  const int kb = (lane < 16) ? 0 : 16;
  return *(const v16bf*)(W + (size_t)n * ldk + (size_t)(k + kb)); // 32B aligned
}

__device__ __forceinline__ float sigmoidf_(float x) {
  return 1.0f / (1.0f + expf(-x));
}

// ---------------------------------------------------------------------------
// fp32 -> bf16 conversion (one-time for x, W_ih, W_hh, W_d, h0, c0)
// ---------------------------------------------------------------------------
__global__ void cvt_f32_to_bf16_kernel(const float* __restrict__ src,
                                       bf16_t* __restrict__ dst, int n) {
  int i = blockIdx.x * blockDim.x + threadIdx.x;
  int stride = gridDim.x * blockDim.x;
  for (; i < n; i += stride) dst[i] = (bf16_t)src[i];
}

// ---------------------------------------------------------------------------
// Time-parallel input projection:
//   Xg[T*B, 4H] = x_bf16[T*B, D] @ W_ih_bf16[4H, D]^T + b_ih + b_hh
// Block = 8 waves, block tile 128(M) x 128(N); each wave: 16(M) x 128(N),
// A-fragment reused across 8 WMMAs per K-step.
// ---------------------------------------------------------------------------
__global__ __launch_bounds__(256) void xg_gemm_kernel(
    const bf16_t* __restrict__ xb,   // [T*B, D]
    const bf16_t* __restrict__ wih,  // [NG, D] row-major
    const float*  __restrict__ b_ih,
    const float*  __restrict__ b_hh,
    float* __restrict__ xg)          // [T*B, NG]
{
  const int lane = threadIdx.x & 31;
  const int wave = threadIdx.x >> 5;
  const int m0 = blockIdx.x * 128 + wave * 16;
  const int n0 = blockIdx.y * 128;

  v8f acc[8] = {};

  const bf16_t* arow = xb + (size_t)(m0 + (lane & 15)) * DIM;

  for (int k = 0; k < DIM; k += 32) {
    v16bf a = load_a_bf16(arow + k, lane);
#pragma unroll
    for (int j = 0; j < 8; ++j) {
      v16bf b = load_b_bf16(wih, n0 + 16 * j, DIM, k, lane);
      acc[j] = __builtin_amdgcn_wmma_f32_16x16x32_bf16(
          false, a, false, b, (short)0, acc[j], false, false);
    }
  }

  const int nl   = lane & 15;
  const int rofs = (lane < 16) ? 0 : 8;
#pragma unroll
  for (int j = 0; j < 8; ++j) {
    const int n = n0 + 16 * j + nl;
    const float bias = b_ih[n] + b_hh[n];
#pragma unroll
    for (int r = 0; r < 8; ++r) {
      const int m = m0 + r + rofs;
      xg[(size_t)m * NG + n] = acc[j][r] + bias;
    }
  }
}

// ---------------------------------------------------------------------------
// One recurrent step (launched T times, newest-to-oldest).
// Recurrent state kept natively in bf16 ping-pong buffers so A-fragments are
// pure 16B loads (no cvt in the K-loop); fp32 c shadow kept for c* and the
// final c output. fp32 h is never materialized: outs[t] IS h_new, and the
// final-h output equals outs[0].
//   C_S    = tanh(c_in @ W_d^T + b_d)
//   c*     = (c_in - C_S) + C_S * decay            decay = 1/ln(e + dt)
//   gates  = Xg[t] + h_in @ W_hh^T                 (i,f,g,o)
//   c_out  = sig(f)*c* + sig(i)*tanh(g)
//   h_out  = sig(o)*tanh(c_out);  outs[t] = h_out
// Block owns a 16-wide H-slice; wave w owns batch rows 16w..16w+15 and runs
// all 5 GEMM pieces so the epilogue is block-local. W_hh/W_d stay hot in L2.
// ---------------------------------------------------------------------------
__global__ __launch_bounds__(128) void tlstm_step_kernel(
    const float*  __restrict__ xg_t,    // [B, NG] slice for this t
    const bf16_t* __restrict__ h_in16,  // [B, H] bf16 state
    const bf16_t* __restrict__ c_in16,  // [B, H] bf16 state
    const float*  __restrict__ c_in,    // [B, H] fp32 state
    bf16_t*       __restrict__ h_out16, // [B, H]
    bf16_t*       __restrict__ c_out16, // [B, H]
    float*        __restrict__ c_out,   // [B, H]
    float*        __restrict__ outs_t,  // [B, H] slice of d_out (= h_new)
    const bf16_t* __restrict__ whh,     // [NG, H]
    const bf16_t* __restrict__ wd,      // [H, H]
    const float*  __restrict__ b_d,     // [H]
    const float*  __restrict__ dts)     // [B]
{
  const int lane = threadIdx.x & 31;
  const int wave = threadIdx.x >> 5;  // 0..3 -> batch-row tile
  const int m0 = wave * 16;
  const int j0 = blockIdx.x * 16;     // H column slice

  v8f acc_cs = {}, acc_i = {}, acc_f = {}, acc_g = {}, acc_o = {};

  const bf16_t* hrow = h_in16 + (size_t)(m0 + (lane & 15)) * HID;
  const bf16_t* crow = c_in16 + (size_t)(m0 + (lane & 15)) * HID;

  for (int k = 0; k < HID; k += 32) {
    v16bf ha = load_a_bf16(hrow + k, lane);
    v16bf ca = load_a_bf16(crow + k, lane);
    v16bf bd  = load_b_bf16(wd,            j0, HID, k, lane);
    v16bf bwi = load_b_bf16(whh, 0 * HID + j0, HID, k, lane);
    v16bf bwf = load_b_bf16(whh, 1 * HID + j0, HID, k, lane);
    v16bf bwg = load_b_bf16(whh, 2 * HID + j0, HID, k, lane);
    v16bf bwo = load_b_bf16(whh, 3 * HID + j0, HID, k, lane);
    acc_cs = __builtin_amdgcn_wmma_f32_16x16x32_bf16(false, ca, false, bd,  (short)0, acc_cs, false, false);
    acc_i  = __builtin_amdgcn_wmma_f32_16x16x32_bf16(false, ha, false, bwi, (short)0, acc_i,  false, false);
    acc_f  = __builtin_amdgcn_wmma_f32_16x16x32_bf16(false, ha, false, bwf, (short)0, acc_f,  false, false);
    acc_g  = __builtin_amdgcn_wmma_f32_16x16x32_bf16(false, ha, false, bwg, (short)0, acc_g,  false, false);
    acc_o  = __builtin_amdgcn_wmma_f32_16x16x32_bf16(false, ha, false, bwo, (short)0, acc_o,  false, false);
  }

  const int nl   = lane & 15;
  const int rofs = (lane < 16) ? 0 : 8;
  const int j    = j0 + nl;
  const float bdj = b_d[j];
#pragma unroll
  for (int r = 0; r < 8; ++r) {
    const int b = m0 + r + rofs;
    const float dt    = dts[b];
    const float decay = 1.0f / logf(2.7182818284590452f + dt);
    const float cprev = c_in[(size_t)b * HID + j];
    const float cs    = tanhf(acc_cs[r] + bdj);
    const float cstar = (cprev - cs) + cs * decay;
    const float gi = sigmoidf_(acc_i[r] + xg_t[(size_t)b * NG + 0 * HID + j]);
    const float gf = sigmoidf_(acc_f[r] + xg_t[(size_t)b * NG + 1 * HID + j]);
    const float gg = tanhf    (acc_g[r] + xg_t[(size_t)b * NG + 2 * HID + j]);
    const float go = sigmoidf_(acc_o[r] + xg_t[(size_t)b * NG + 3 * HID + j]);
    const float cn = gf * cstar + gi * gg;
    const float hn = go * tanhf(cn);
    const size_t idx = (size_t)b * HID + j;
    c_out  [idx] = cn;
    c_out16[idx] = (bf16_t)cn;
    h_out16[idx] = (bf16_t)hn;
    outs_t [idx] = hn;
  }
}

// ---------------------------------------------------------------------------
extern "C" void kernel_launch(void* const* d_in, const int* in_sizes, int n_in,
                              void* d_out, int out_size, void* d_ws, size_t ws_size,
                              hipStream_t stream) {
  (void)in_sizes; (void)n_in; (void)out_size; (void)ws_size;

  const float* x    = (const float*)d_in[0]; // [T,B,D]
  const float* h0   = (const float*)d_in[1]; // [B,H]
  const float* c0   = (const float*)d_in[2]; // [B,H]
  const float* dts  = (const float*)d_in[3]; // [B]
  const float* wih  = (const float*)d_in[4]; // [4H,D]
  const float* whh  = (const float*)d_in[5]; // [4H,H]
  const float* bih  = (const float*)d_in[6]; // [4H]
  const float* bhh  = (const float*)d_in[7]; // [4H]
  const float* wd   = (const float*)d_in[8]; // [H,H]
  const float* bd   = (const float*)d_in[9]; // [H]
  float* out = (float*)d_out;                // outs[T,B,H] ++ h[B,H] ++ c[B,H]

  // Workspace carve-out (~595 MB): Xg fp32, bf16 copies, state ping-pong.
  char* ws = (char*)d_ws;
  size_t off = 0;
  float*  xg   = (float*) (ws + off); off += (size_t)T_LEN * BATCH * NG * sizeof(float);
  bf16_t* xb   = (bf16_t*)(ws + off); off += (size_t)T_LEN * BATCH * DIM * sizeof(bf16_t);
  bf16_t* wihb = (bf16_t*)(ws + off); off += (size_t)NG * DIM * sizeof(bf16_t);
  bf16_t* whhb = (bf16_t*)(ws + off); off += (size_t)NG * HID * sizeof(bf16_t);
  bf16_t* wdb  = (bf16_t*)(ws + off); off += (size_t)HID * HID * sizeof(bf16_t);
  bf16_t* hb16[2]; bf16_t* cb16[2]; float* cb[2];
  hb16[0] = (bf16_t*)(ws + off); off += (size_t)BATCH * HID * sizeof(bf16_t);
  hb16[1] = (bf16_t*)(ws + off); off += (size_t)BATCH * HID * sizeof(bf16_t);
  cb16[0] = (bf16_t*)(ws + off); off += (size_t)BATCH * HID * sizeof(bf16_t);
  cb16[1] = (bf16_t*)(ws + off); off += (size_t)BATCH * HID * sizeof(bf16_t);
  cb[0]   = (float*) (ws + off); off += (size_t)BATCH * HID * sizeof(float);
  cb[1]   = (float*) (ws + off); off += (size_t)BATCH * HID * sizeof(float);

  // One-time fp32 -> bf16 conversions
  cvt_f32_to_bf16_kernel<<<4096, 256, 0, stream>>>(x,   xb,   T_LEN * BATCH * DIM);
  cvt_f32_to_bf16_kernel<<<512,  256, 0, stream>>>(wih, wihb, NG * DIM);
  cvt_f32_to_bf16_kernel<<<512,  256, 0, stream>>>(whh, whhb, NG * HID);
  cvt_f32_to_bf16_kernel<<<256,  256, 0, stream>>>(wd,  wdb,  HID * HID);
  cvt_f32_to_bf16_kernel<<<64,   256, 0, stream>>>(h0,  hb16[0], BATCH * HID);
  cvt_f32_to_bf16_kernel<<<64,   256, 0, stream>>>(c0,  cb16[0], BATCH * HID);

  // Time-parallel input projection (~44% of the GEMM FLOPs)
  dim3 ggrid((T_LEN * BATCH) / 128, NG / 128);
  xg_gemm_kernel<<<ggrid, 256, 0, stream>>>(xb, wihb, bih, bhh, xg);

  // Seed fp32 c state
  hipMemcpyAsync(cb[0], c0, (size_t)BATCH * HID * sizeof(float), hipMemcpyDeviceToDevice, stream);

  // Reverse-time recurrence: t = T-1 .. 0, ping-pong state buffers.
  int cur = 0;
  for (int t = T_LEN - 1; t >= 0; --t) {
    const int nxt = cur ^ 1;
    tlstm_step_kernel<<<HID / 16, 128, 0, stream>>>(
        xg + (size_t)t * BATCH * NG,
        hb16[cur], cb16[cur], cb[cur],
        hb16[nxt], cb16[nxt], cb[nxt],
        out + (size_t)t * BATCH * HID,
        whhb, wdb, bd, dts);
    cur = nxt;
  }

  // Final h = outs[0] (the last h_new of the reverse scan); final c from state.
  hipMemcpyAsync(out + (size_t)T_LEN * BATCH * HID,
                 out, (size_t)BATCH * HID * sizeof(float), hipMemcpyDeviceToDevice, stream);
  hipMemcpyAsync(out + (size_t)T_LEN * BATCH * HID + (size_t)BATCH * HID,
                 cb[cur], (size_t)BATCH * HID * sizeof(float), hipMemcpyDeviceToDevice, stream);
}
